// NoteAxis_335007449484
// MI455X (gfx1250) — compile-verified, matrix operand order checked by reading, not run on hardware
//
#include <hip/hip_runtime.h>
#include <hip/hip_bf16.h>

// ---------------- problem constants ----------------
#define BATCH 512
#define NSTEP 48
#define NFEAT 128
#define UNITS 512
#define G4    2048          // 4 * UNITS (gate rows)
#define K0    672           // padded input0: 129 -> 160, + 512 hidden
#define K1    1024          // hidden0 (512) + hidden1 (512)
#define BT    16            // batch tile per workgroup
#define S0    680           // LDS stride (halves) for in0, padded (bank-safe)
#define S1    1032          // LDS stride (halves) for in1, padded (bank-safe)
#define NTHR  512           // 16 waves per block

typedef __attribute__((ext_vector_type(16))) _Float16 v16h;
typedef __attribute__((ext_vector_type(8)))  float    v8f;
typedef __attribute__((ext_vector_type(4)))  unsigned int v4u;

// ws layout (bytes)
#define WA_BYTES  ((size_t)G4 * K0 * 2)                 // 2,752,512
#define WB_BYTES  ((size_t)G4 * K1 * 2)                 // 4,194,304
#define WA_OFF    ((size_t)0)
#define WB_OFF    (WA_BYTES)
#define SEQ_OFF   (WA_BYTES + WB_BYTES)

static __device__ __forceinline__ float sigm(float x) {
    return 1.0f / (1.0f + __expf(-x));
}
static __device__ __forceinline__ float tanh_fast(float x) {
    return 1.0f - 2.0f / (1.0f + __expf(2.0f * x));
}

// Load a 16x32 f16 A/B operand tile: per-lane two contiguous 16B chunks,
// matching the CDNA5 16-bit 16x32 operand VGPR layout (lane-half K split).
static __device__ __forceinline__ v16h load_tile(const _Float16* p) {
    union { v16h h; v4u u[2]; } t;
    t.u[0] = *(const v4u*)(p);       // K = c .. c+7
    t.u[1] = *(const v4u*)(p + 16);  // K = c+16 .. c+23
    return t.h;
}

// Software-pipelined 8-tile GEMM slice: this wave owns gate rows
//   row(m) = (m>>1)*512 + u0 + (m&1)*16,  m = 0..7
// A (weights) streams from global/L2, B (activations) from LDS.
template <int KS, int KT>
static __device__ __forceinline__ void gemm_8tiles(const _Float16* __restrict__ wlane,
                                                   const _Float16* actrow,  // &in[bl][hi8]
                                                   v8f acc[8]) {
    v16h A[4], Bt[4], bt;
    // preload group A (tiles 0..3) for kt = 0
    #pragma unroll
    for (int m = 0; m < 4; ++m)
        A[m] = load_tile(wlane + (size_t)((m >> 1) * 512 + (m & 1) * 16) * KS);

    #pragma unroll 1
    for (int kt = 0; kt < KT; ++kt) {
        bt = load_tile(actrow + kt * 32);
        // issue group-B loads (tiles 4..7) for this kt
        #pragma unroll
        for (int m = 0; m < 4; ++m)
            Bt[m] = load_tile(wlane +
                (size_t)(((m + 4) >> 1) * 512 + ((m + 4) & 1) * 16) * KS + kt * 32);
        // WMMA group A (operands loaded in previous iteration) — overlaps GB loads
        #pragma unroll
        for (int m = 0; m < 4; ++m)
            acc[m] = __builtin_amdgcn_wmma_f32_16x16x32_f16(
                false, A[m], false, bt, (short)0, acc[m], false, false);
        // issue group-A loads for kt+1 — overlap with WMMA group B below
        if (kt + 1 < KT) {
            #pragma unroll
            for (int m = 0; m < 4; ++m)
                A[m] = load_tile(wlane +
                    (size_t)((m >> 1) * 512 + (m & 1) * 16) * KS + (kt + 1) * 32);
        }
        // WMMA group B
        #pragma unroll
        for (int m = 0; m < 4; ++m)
            acc[m + 4] = __builtin_amdgcn_wmma_f32_16x16x32_f16(
                false, Bt[m], false, bt, (short)0, acc[m + 4], false, false);
    }
}

// ---------------- prep kernels ----------------
__global__ void pack_wa_kernel(const float* __restrict__ Wih0,
                               const float* __restrict__ Whh0,
                               _Float16* __restrict__ WA) {
    int idx = blockIdx.x * 256 + threadIdx.x;
    if (idx >= G4 * K0) return;
    int g = idx / K0, k = idx % K0;
    float v = 0.0f;
    if (k < 129)       v = Wih0[g * 129 + k];
    else if (k >= 160) v = Whh0[g * UNITS + (k - 160)];
    WA[idx] = (_Float16)v;
}

__global__ void pack_wb_kernel(const float* __restrict__ Wih1,
                               const float* __restrict__ Whh1,
                               _Float16* __restrict__ WB) {
    int idx = blockIdx.x * 256 + threadIdx.x;
    if (idx >= G4 * K1) return;
    int g = idx / K1, k = idx % K1;
    float v = (k < UNITS) ? Wih1[g * UNITS + k] : Whh1[g * UNITS + (k - UNITS)];
    WB[idx] = (_Float16)v;
}

__global__ void pack_seq_kernel(const float* __restrict__ nf,
                                const float* __restrict__ tg,
                                _Float16* __restrict__ seq) {
    int idx = blockIdx.x * 256 + threadIdx.x;
    if (idx >= BATCH * NSTEP * 160) return;
    int b = idx / (NSTEP * 160);
    int r = idx % (NSTEP * 160);
    int n = r / 160, k = r % 160;
    float v = 0.0f;
    if (k < NFEAT)                 v = nf[((size_t)b * NSTEP + n) * NFEAT + k];
    else if (k == NFEAT && n > 0)  v = tg[b * NSTEP + (n - 1)];
    seq[idx] = (_Float16)v;
}

// ---------------- main fused LSTM kernel ----------------
__global__ __launch_bounds__(NTHR, 1)
void noteaxis_lstm_kernel(const _Float16* __restrict__ WA,
                          const _Float16* __restrict__ WB,
                          const _Float16* __restrict__ seq,
                          const float* __restrict__ b_ih0, const float* __restrict__ b_hh0,
                          const float* __restrict__ b_ih1, const float* __restrict__ b_hh1,
                          const float* __restrict__ W_out, const float* __restrict__ b_out,
                          float* __restrict__ out) {
    __shared__ __align__(16) _Float16 in0[BT][S0];      // [x(160) | h0(512)] per batch row
    __shared__ __align__(16) _Float16 in1[BT][S1];      // [h0(512) | h1(512)] per batch row
    __shared__ __align__(16) _Float16 biasl[2][G4];     // summed gate biases, f16
    __shared__ __align__(16) float    woutl[UNITS];     // output weights, f32

    const int tid  = threadIdx.x;
    const int lane = tid & 31;
    const int wave = tid >> 5;            // 0..15
    const int bl   = lane & 15;           // batch column / weight row within tile
    const int hi8  = (lane >> 4) * 8;     // K-half select for 16-bit operands
    const int u0   = wave * 32;           // this wave's hidden-unit range (32 units)
    const int b0   = blockIdx.x * BT;     // batch base

    // one-time LDS init
    for (int i = tid; i < G4; i += NTHR) {
        biasl[0][i] = (_Float16)(b_ih0[i] + b_hh0[i]);
        biasl[1][i] = (_Float16)(b_ih1[i] + b_hh1[i]);
    }
    for (int i = tid; i < UNITS; i += NTHR) woutl[i] = W_out[i];
    for (int i = tid; i < BT * S0; i += NTHR) ((_Float16*)in0)[i] = (_Float16)0.0f;
    for (int i = tid; i < BT * S1; i += NTHR) ((_Float16*)in1)[i] = (_Float16)0.0f;

    float c0[2][8], c1[2][8];
    #pragma unroll
    for (int j = 0; j < 2; ++j)
        #pragma unroll
        for (int r = 0; r < 8; ++r) { c0[j][r] = 0.0f; c1[j][r] = 0.0f; }

    const _Float16* wa_lane = WA + (size_t)(u0 + bl) * K0 + hi8;
    const _Float16* wb_lane = WB + (size_t)(u0 + bl) * K1 + hi8;

    __syncthreads();

    for (int n = 0; n < NSTEP; ++n) {
        // ---- stage x_t into in0 via CDNA5 async global->LDS copies ----
        {
            int c = tid;                     // 320 chunks, NTHR=512 threads: one each
            if (c < BT * 20) {
                int row = c / 20, cc = (c % 20) * 8;
                const _Float16* g = seq + ((size_t)(b0 + row) * NSTEP + n) * 160 + cc;
                unsigned           ldsa = (unsigned)(uintptr_t)&in0[row][cc];
                unsigned long long ga   = (unsigned long long)(uintptr_t)g;
                asm volatile("global_load_async_to_lds_b128 %0, %1, off"
                             :: "v"(ldsa), "v"(ga) : "memory");
            }
            asm volatile("s_wait_asynccnt 0x0" ::: "memory");
        }
        __syncthreads();

        // ================= layer 0: gates0 = [Wih0|Whh0] x [x|h0] =================
        v8f acc[8];
        #pragma unroll
        for (int g = 0; g < 4; ++g)
            #pragma unroll
            for (int j = 0; j < 2; ++j)
                #pragma unroll
                for (int r = 0; r < 8; ++r)
                    acc[g * 2 + j][r] =
                        (float)biasl[0][g * UNITS + u0 + j * 16 + r + hi8];

        gemm_8tiles<K0, K0 / 32>(wa_lane, &in0[bl][hi8], acc);
        __syncthreads();   // all waves done reading in0

        // ---- layer 0 pointwise: c0,h0 update; h0 -> in0[:,160+u] and in1[:,u] ----
        #pragma unroll
        for (int j = 0; j < 2; ++j)
            #pragma unroll
            for (int r = 0; r < 8; ++r) {
                float iv = sigm(acc[0 * 2 + j][r]);
                float fv = sigm(acc[1 * 2 + j][r]);
                float gv = tanh_fast(acc[2 * 2 + j][r]);
                float ov = sigm(acc[3 * 2 + j][r]);
                float c  = fv * c0[j][r] + iv * gv;
                c0[j][r] = c;
                _Float16 h = (_Float16)(ov * tanh_fast(c));
                int u = u0 + j * 16 + r + hi8;
                in0[bl][160 + u] = h;
                in1[bl][u]       = h;
            }
        __syncthreads();

        // ================= layer 1: gates1 = [Wih1|Whh1] x [h0|h1] =================
        #pragma unroll
        for (int g = 0; g < 4; ++g)
            #pragma unroll
            for (int j = 0; j < 2; ++j)
                #pragma unroll
                for (int r = 0; r < 8; ++r)
                    acc[g * 2 + j][r] =
                        (float)biasl[1][g * UNITS + u0 + j * 16 + r + hi8];

        gemm_8tiles<K1, K1 / 32>(wb_lane, &in1[bl][hi8], acc);
        __syncthreads();   // all waves done reading in1

        // ---- layer 1 pointwise: c1,h1 update; h1 -> in1[:,512+u] ----
        #pragma unroll
        for (int j = 0; j < 2; ++j)
            #pragma unroll
            for (int r = 0; r < 8; ++r) {
                float iv = sigm(acc[0 * 2 + j][r]);
                float fv = sigm(acc[1 * 2 + j][r]);
                float gv = tanh_fast(acc[2 * 2 + j][r]);
                float ov = sigm(acc[3 * 2 + j][r]);
                float c  = fv * c1[j][r] + iv * gv;
                c1[j][r] = c;
                in1[bl][UNITS + u0 + j * 16 + r + hi8] = (_Float16)(ov * tanh_fast(c));
            }
        __syncthreads();

        // ---- output head: y[b,n] = sigmoid(h1 . W_out + b_out), wave 0 only ----
        if (wave == 0) {
            int b = lane & 15, hh = lane >> 4;
            float s = 0.0f;
            for (int u = hh * 256; u < hh * 256 + 256; u += 8) {
                union { v4u v; _Float16 h[8]; } t;
                t.v = *(const v4u*)&in1[b][UNITS + u];
                #pragma unroll
                for (int q = 0; q < 8; ++q) s += (float)t.h[q] * woutl[u + q];
            }
            s += __shfl_xor(s, 16, 32);      // combine the two 256-unit halves
            if (hh == 0)
                out[(size_t)(b0 + b) * NSTEP + n] = sigm(s + b_out[0]);
        }
        __syncthreads();
    }
}

// ---------------- host launcher ----------------
extern "C" void kernel_launch(void* const* d_in, const int* in_sizes, int n_in,
                              void* d_out, int out_size, void* d_ws, size_t ws_size,
                              hipStream_t stream) {
    (void)in_sizes; (void)n_in; (void)out_size; (void)ws_size;
    const float* note_features = (const float*)d_in[0];
    const float* targets       = (const float*)d_in[1];
    const float* W_ih0 = (const float*)d_in[2];
    const float* W_hh0 = (const float*)d_in[3];
    const float* b_ih0 = (const float*)d_in[4];
    const float* b_hh0 = (const float*)d_in[5];
    const float* W_ih1 = (const float*)d_in[6];
    const float* W_hh1 = (const float*)d_in[7];
    const float* b_ih1 = (const float*)d_in[8];
    const float* b_hh1 = (const float*)d_in[9];
    const float* W_out = (const float*)d_in[10];
    const float* b_out = (const float*)d_in[11];
    float* out = (float*)d_out;

    _Float16* WA  = (_Float16*)((char*)d_ws + WA_OFF);
    _Float16* WB  = (_Float16*)((char*)d_ws + WB_OFF);
    _Float16* seq = (_Float16*)((char*)d_ws + SEQ_OFF);

    pack_wa_kernel<<<(G4 * K0 + 255) / 256, 256, 0, stream>>>(W_ih0, W_hh0, WA);
    pack_wb_kernel<<<(G4 * K1 + 255) / 256, 256, 0, stream>>>(W_ih1, W_hh1, WB);
    pack_seq_kernel<<<(BATCH * NSTEP * 160 + 255) / 256, 256, 0, stream>>>(
        note_features, targets, seq);

    noteaxis_lstm_kernel<<<BATCH / BT, NTHR, 0, stream>>>(
        WA, WB, seq, b_ih0, b_hh0, b_ih1, b_hh1, W_out, b_out, out);
}
